// GCN_50680614092807
// MI455X (gfx1250) — matrix operand order repaired and checked by reading, not compiled, not run on hardware
//
#include <hip/hip_runtime.h>
#include <hip/hip_bf16.h>

typedef __attribute__((ext_vector_type(16))) _Float16 v16h;
typedef __attribute__((ext_vector_type(8)))  float    v8f;

#define NNODES   100000
#define NEDGES   3200000
#define FIN      165
#define HID      16
#define NCLS     2

// ---------------------------------------------------------------------------
// Zero-fill workspace region
// ---------------------------------------------------------------------------
__global__ void zero_kernel(float* __restrict__ p, long long count) {
    long long i = (long long)blockIdx.x * blockDim.x + threadIdx.x;
    if (i < count) p[i] = 0.0f;
}

// ---------------------------------------------------------------------------
// deg[dst] += 1 per edge (deg accumulated into dinv buffer)
// ---------------------------------------------------------------------------
__global__ void deg_kernel(const int* __restrict__ ei, float* __restrict__ deg, int E_) {
    long long gid = (long long)blockIdx.x * blockDim.x + threadIdx.x;
    if (gid >= E_) return;
    int d = ei[E_ + (int)gid];            // edge_index[1][e]
    atomicAdd(&deg[d], 1.0f);
}

// dinv[i] = rsqrt(deg[i] + 1)   (self-loop)
__global__ void dinv_kernel(float* __restrict__ dinv, int n) {
    int i = blockIdx.x * blockDim.x + threadIdx.x;
    if (i >= n) return;
    dinv[i] = rsqrtf(dinv[i] + 1.0f);
}

// ---------------------------------------------------------------------------
// h = x @ W1 via V_WMMA_F32_16X16X32_F16.
// Block = 128 threads = 4 waves; each wave owns a 16-row tile (block: 64 rows).
// Full-tile blocks stage x through LDS with global_load_async_to_lds_b32;
// the single ragged block uses guarded loads.
// ---------------------------------------------------------------------------
template <bool FULL>
__device__ __forceinline__ void gemm1_body(const float* __restrict__ x,
                                           const float* __restrict__ W1,
                                           float* __restrict__ h, int n,
                                           float (*xt)[33]) {
    const int t        = threadIdx.x;     // 0..127
    const int wave     = t >> 5;          // 0..3
    const int lane     = t & 31;
    const int half     = lane >> 4;       // 0 or 1
    const int l16      = lane & 15;
    const int blockRow = blockIdx.x * 64;
    const int waveRow  = blockRow + wave * 16;

    v8f c = {};

    #pragma unroll
    for (int k0 = 0; k0 < FIN; k0 += 32) {   // 6 K-steps (5 full + 5-col tail)
        __syncthreads();
        if (FULL && (k0 + 32 <= FIN)) {
            // Direct memory -> LDS async copy (no VGPR round trip).
            #pragma unroll
            for (int idx = t; idx < 64 * 32; idx += 128) {
                int r  = idx >> 5;
                int cc = idx & 31;
                unsigned       loff = (unsigned)(unsigned long long)(uintptr_t)&xt[r][cc];
                unsigned long long gaddr =
                    (unsigned long long)(uintptr_t)(x + (long long)(blockRow + r) * FIN + k0 + cc);
                asm volatile("global_load_async_to_lds_b32 %0, %1, off"
                             :: "v"(loff), "v"(gaddr) : "memory");
            }
            asm volatile("s_wait_asynccnt 0x0" ::: "memory");
        } else {
            // Guarded path (row tail block and/or K tail step).
            for (int idx = t; idx < 64 * 32; idx += 128) {
                int r  = idx >> 5;
                int cc = idx & 31;
                int gr = blockRow + r;
                int gk = k0 + cc;
                bool ok = (FULL || gr < n) && (gk < FIN);
                xt[r][cc] = ok ? x[(long long)gr * FIN + gk] : 0.0f;
            }
        }
        __syncthreads();

        // A fragment: 16-bit A 16x32 layout.
        // half 0: V0..3 -> K 0..7, V4..7 -> K 16..23 ; half 1: +8 (and +24)
        v16h a;
        const int arow = wave * 16 + l16;
        #pragma unroll
        for (int j = 0; j < 16; ++j) {
            int v = j >> 1;
            int k = ((v >> 2) << 4) + (half << 3) + ((v & 3) << 1) + (j & 1);
            a[j] = (_Float16)xt[arow][k];
        }

        // B fragment: 32x16, lanes 0-15 hold K=0..7,16..23, lanes 16-31 hold K=8..15,24..31
        v16h b;
        #pragma unroll
        for (int j = 0; j < 16; ++j) {
            int k = k0 + ((j >> 3) << 4) + (half << 3) + (j & 7);
            b[j] = (k < FIN) ? (_Float16)W1[k * HID + l16] : (_Float16)0.0f;
        }

        c = __builtin_amdgcn_wmma_f32_16x16x32_f16(
                /*neg_a=*/false, a, /*neg_b=*/false, b,
                /*c_mod=*/(short)0, c, /*reuse_a=*/false, /*reuse_b=*/false);
    }

    // C/D layout: VGPR r -> rows (r, r+8) for lane halves (0,1); col = lane&15
    #pragma unroll
    for (int r = 0; r < 8; ++r) {
        int grow = waveRow + r + (half << 3);
        if (FULL || grow < n) h[(long long)grow * HID + l16] = c[r];
    }
}

__global__ void __launch_bounds__(128)
gemm1_wmma_kernel(const float* __restrict__ x, const float* __restrict__ W1,
                  float* __restrict__ h, int n) {
    __shared__ float xt[64][33];          // pad stride 33 to avoid bank conflicts
    if (blockIdx.x * 64 + 64 <= n) {
        gemm1_body<true>(x, W1, h, n, xt);
    } else {
        gemm1_body<false>(x, W1, h, n, xt);
    }
}

// ---------------------------------------------------------------------------
// agg1[dst] += dinv[src]*dinv[dst] * h[src]   (4 threads per edge, float4 each)
// ---------------------------------------------------------------------------
__global__ void prop1_kernel(const int* __restrict__ ei, const float* __restrict__ h,
                             const float* __restrict__ dinv, float* __restrict__ agg,
                             int E_) {
    long long gid = (long long)blockIdx.x * blockDim.x + threadIdx.x;
    int e = (int)(gid >> 2);
    int q = (int)(gid & 3);
    if (e >= E_) return;
    int s = ei[e];
    int d = ei[E_ + e];
    float w = dinv[s] * dinv[d];
    const float4 hv = *(const float4*)(h + (long long)s * HID + q * 4);
    float* ap = agg + (long long)d * HID + q * 4;
    atomicAdd(ap + 0, w * hv.x);
    atomicAdd(ap + 1, w * hv.y);
    atomicAdd(ap + 2, w * hv.z);
    atomicAdd(ap + 3, w * hv.w);
}

// ---------------------------------------------------------------------------
// h2 = relu(agg1 + dinv^2 * h + b1) @ W2      (per node)
// ---------------------------------------------------------------------------
__global__ void layer2_kernel(const float* __restrict__ agg1, const float* __restrict__ h,
                              const float* __restrict__ dinv, const float* __restrict__ b1,
                              const float* __restrict__ W2, float* __restrict__ h2, int n) {
    int i = blockIdx.x * blockDim.x + threadIdx.x;
    if (i >= n) return;
    float di2 = dinv[i] * dinv[i];
    float o0 = 0.0f, o1 = 0.0f;
    #pragma unroll
    for (int cidx = 0; cidx < HID; ++cidx) {
        float v = agg1[(long long)i * HID + cidx] + di2 * h[(long long)i * HID + cidx] + b1[cidx];
        v = fmaxf(v, 0.0f);
        o0 += v * W2[cidx * NCLS + 0];
        o1 += v * W2[cidx * NCLS + 1];
    }
    h2[(long long)i * NCLS + 0] = o0;
    h2[(long long)i * NCLS + 1] = o1;
}

// ---------------------------------------------------------------------------
// agg2[dst] += dinv[src]*dinv[dst] * h2[src]   (1 thread per edge, D=2)
// ---------------------------------------------------------------------------
__global__ void prop2_kernel(const int* __restrict__ ei, const float* __restrict__ h2,
                             const float* __restrict__ dinv, float* __restrict__ agg2,
                             int E_) {
    long long gid = (long long)blockIdx.x * blockDim.x + threadIdx.x;
    if (gid >= E_) return;
    int e = (int)gid;
    int s = ei[e];
    int d = ei[E_ + e];
    float w = dinv[s] * dinv[d];
    const float2 v = *(const float2*)(h2 + (long long)s * NCLS);
    atomicAdd(agg2 + (long long)d * NCLS + 0, w * v.x);
    atomicAdd(agg2 + (long long)d * NCLS + 1, w * v.y);
}

// ---------------------------------------------------------------------------
// out = log_softmax(agg2 + dinv^2 * h2 + b2)
// ---------------------------------------------------------------------------
__global__ void final_kernel(const float* __restrict__ agg2, const float* __restrict__ h2,
                             const float* __restrict__ dinv, const float* __restrict__ b2,
                             float* __restrict__ out, int n) {
    int i = blockIdx.x * blockDim.x + threadIdx.x;
    if (i >= n) return;
    float di2 = dinv[i] * dinv[i];
    float z0 = agg2[(long long)i * NCLS + 0] + di2 * h2[(long long)i * NCLS + 0] + b2[0];
    float z1 = agg2[(long long)i * NCLS + 1] + di2 * h2[(long long)i * NCLS + 1] + b2[1];
    float m  = fmaxf(z0, z1);
    float l  = m + logf(expf(z0 - m) + expf(z1 - m));
    out[(long long)i * NCLS + 0] = z0 - l;
    out[(long long)i * NCLS + 1] = z1 - l;
}

// ---------------------------------------------------------------------------
// Launcher
// ---------------------------------------------------------------------------
extern "C" void kernel_launch(void* const* d_in, const int* in_sizes, int n_in,
                              void* d_out, int out_size, void* d_ws, size_t ws_size,
                              hipStream_t stream) {
    const float* x  = (const float*)d_in[0];
    const int*   ei = (const int*)  d_in[1];   // [2,E] flat: src = ei[e], dst = ei[E+e]
    const float* W1 = (const float*)d_in[2];
    const float* b1 = (const float*)d_in[3];
    const float* W2 = (const float*)d_in[4];
    const float* b2 = (const float*)d_in[5];
    float* out = (float*)d_out;

    const int n = NNODES;
    const int E = NEDGES;

    // workspace layout (floats): [dinv(N) | agg1(16N) | agg2(2N) | h(16N) | h2(2N)]
    float* ws   = (float*)d_ws;
    float* dinv = ws;
    float* agg1 = ws + (long long)n;
    float* agg2 = ws + (long long)n * 17;
    float* h    = ws + (long long)n * 19;
    float* h2   = ws + (long long)n * 35;

    const int B = 256;

    // 1) zero dinv + agg1 + agg2 (19N floats)
    {
        long long cnt = (long long)n * 19;
        int blocks = (int)((cnt + B - 1) / B);
        zero_kernel<<<blocks, B, 0, stream>>>(ws, cnt);
    }
    // 2) degree count
    deg_kernel<<<(E + B - 1) / B, B, 0, stream>>>(ei, dinv, E);
    // 3) dinv = rsqrt(deg + 1)
    dinv_kernel<<<(n + B - 1) / B, B, 0, stream>>>(dinv, n);
    // 4) h = x @ W1  (WMMA, async LDS staging)
    {
        int blocks = (n + 63) / 64;
        gemm1_wmma_kernel<<<blocks, 128, 0, stream>>>(x, W1, h, n);
    }
    // 5) propagate layer 1 (4 threads/edge)
    {
        long long thr = (long long)E * 4;
        prop1_kernel<<<(int)((thr + B - 1) / B), B, 0, stream>>>(ei, h, dinv, agg1, E);
    }
    // 6) bias + relu + h2 = (.) @ W2
    layer2_kernel<<<(n + B - 1) / B, B, 0, stream>>>(agg1, h, dinv, b1, W2, h2, n);
    // 7) propagate layer 2
    prop2_kernel<<<(E + B - 1) / B, B, 0, stream>>>(ei, h2, dinv, agg2, E);
    // 8) log_softmax
    final_kernel<<<(n + B - 1) / B, B, 0, stream>>>(agg2, h2, dinv, b2, out, n);
}
